// VIDecoder_32263794327961
// MI455X (gfx1250) — matrix operand order repaired and checked by reading, not compiled, not run on hardware
//
#include <hip/hip_runtime.h>

#define SEQ    2048
#define NSTEP  2047
#define CCH    6
#define SIGC   258      // 6 + 36 + 216
#define KPAD   288      // 258 padded to a multiple of 32 (9 WMMA K-steps)
#define OUTC   128
#define NBATCH 64
#define CHUNK  128      // output rows per phase-C block
#define NCHUNK 16
#define NTOT   15       // chunk totals needed for the prefix scan

typedef __attribute__((ext_vector_type(16))) __bf16 v16bf;
typedef __attribute__((ext_vector_type(8)))  __bf16 v8bf;
typedef __attribute__((ext_vector_type(8)))  float  v8f;

static __device__ __forceinline__ __bf16 f2bf(float f) {
  union { float f; unsigned u; } x; x.f = f;
  unsigned u = x.u + 0x7FFFu + ((x.u >> 16) & 1u);   // round-to-nearest-even
  unsigned short s = (unsigned short)(u >> 16);
  __bf16 r;
  __builtin_memcpy(&r, &s, 2);
  return r;
}

// Map flat signature element id -> (level, i, j, k)
static __device__ __forceinline__ void decode_elem(int e, int& lvl, int& i, int& j, int& k) {
  if (e < 6)       { lvl = 1; i = e;  j = 0; k = 0; }
  else if (e < 42) { int ij = e - 6;  lvl = 2; i = ij / 6;  j = ij % 6;        k = 0; }
  else             { int ijk = e - 42; lvl = 3; i = ijk / 36; j = (ijk / 6) % 6; k = ijk % 6; }
}

// One Chen step with a single-segment exponential B = exp(dx)
static __device__ __forceinline__ float chen_step(float a, int lvl, int i, int j, int k,
                                                  const float* dx, const float* A1, const float* A2) {
  float di = dx[i];
  if (lvl == 1) return a + di;
  float dj = dx[j];
  if (lvl == 2) return a + di * dj * 0.5f + A1[i] * dj;
  float dk = dx[k];
  return a + di * dj * dk * (1.0f / 6.0f) + A1[i] * (dj * dk * 0.5f) + A2[i * 6 + j] * dk;
}

// General Chen combine with a full signature B (for the prefix scan)
static __device__ __forceinline__ float chen_comb(float a, int lvl, int i, int j, int k,
                                                  float be, const float* B1, const float* B2,
                                                  const float* A1, const float* A2) {
  if (lvl == 1) return a + be;
  if (lvl == 2) return a + be + A1[i] * B1[j];
  return a + be + A1[i] * B2[j * 6 + k] + A2[i * 6 + j] * B1[k];
}

// ---------------- Phase A: per-chunk signature totals ----------------
__global__ void __launch_bounds__(256) sig_chunk_totals(const float* __restrict__ inp,
                                                        float* __restrict__ ws_total) {
  const int c = blockIdx.x;   // 0..NTOT-1
  const int b = blockIdx.y;   // 0..63
  __shared__ float sDx[CHUNK * CCH];
  __shared__ float sA1[2][6];
  __shared__ float sA2[2][36];
  const int t = threadIdx.x;

  for (int idx = t; idx < CHUNK * CCH; idx += 256) {
    int l = idx / CCH, ch = idx % CCH;
    int s = c * CHUNK + l;                     // dx index, max 14*128+127 = 1919
    float v;
    if (ch == 0) v = 1.0f / (float)NSTEP;      // time channel increment
    else {
      size_t o = ((size_t)b * SEQ + s) * 5 + (ch - 1);
      v = inp[o + 5] - inp[o];
    }
    sDx[idx] = v;
  }
  if (t < 6)  sA1[0][t] = 0.f;
  if (t < 36) sA2[0][t] = 0.f;

  int l0, i0, j0, k0, l1 = 0, i1 = 0, j1 = 0, k1 = 0;
  decode_elem(t, l0, i0, j0, k0);
  if (t < 2) decode_elem(256 + t, l1, i1, j1, k1);
  float a0 = 0.f, a1 = 0.f;
  __syncthreads();

  int p = 0;
  for (int l = 0; l < CHUNK; ++l) {
    const float* dx = &sDx[l * CCH];
    a0 = chen_step(a0, l0, i0, j0, k0, dx, sA1[p], sA2[p]);
    if (t < 2) a1 = chen_step(a1, l1, i1, j1, k1, dx, sA1[p], sA2[p]);
    if (t < 6)       sA1[p ^ 1][t] = a0;
    else if (t < 42) sA2[p ^ 1][t - 6] = a0;
    p ^= 1;
    __syncthreads();
  }
  size_t base = ((size_t)b * NTOT + c) * SIGC;
  ws_total[base + t] = a0;
  if (t < 2) ws_total[base + 256 + t] = a1;
}

// ---------------- Phase B: per-batch exclusive prefix (Chen combine) ----------------
__global__ void __launch_bounds__(256) sig_prefix(const float* __restrict__ ws_total,
                                                  float* __restrict__ ws_prefix) {
  const int b = blockIdx.x;
  __shared__ float sA1[2][6];
  __shared__ float sA2[2][36];
  __shared__ float sB1[6];
  __shared__ float sB2[36];
  const int t = threadIdx.x;

  int l0, i0, j0, k0, l1 = 0, i1 = 0, j1 = 0, k1 = 0;
  decode_elem(t, l0, i0, j0, k0);
  if (t < 2) decode_elem(256 + t, l1, i1, j1, k1);
  float a0 = 0.f, a1 = 0.f;

  size_t pbase = (size_t)b * NCHUNK * SIGC;
  ws_prefix[pbase + t] = 0.f;                 // prefix[0] = zero signature
  if (t < 2) ws_prefix[pbase + 256 + t] = 0.f;

  if (t < 6)  sA1[0][t] = 0.f;
  if (t < 36) sA2[0][t] = 0.f;
  __syncthreads();

  int p = 0;
  for (int c = 0; c < NTOT; ++c) {
    const float* B = &ws_total[((size_t)b * NTOT + c) * SIGC];
    float be0 = B[t];
    float be1 = (t < 2) ? B[256 + t] : 0.f;
    if (t < 6)       sB1[t] = be0;
    else if (t < 42) sB2[t - 6] = be0;
    __syncthreads();
    a0 = chen_comb(a0, l0, i0, j0, k0, be0, sB1, sB2, sA1[p], sA2[p]);
    if (t < 2) a1 = chen_comb(a1, l1, i1, j1, k1, be1, sB1, sB2, sA1[p], sA2[p]);
    if (t < 6)       sA1[p ^ 1][t] = a0;
    else if (t < 42) sA2[p ^ 1][t - 6] = a0;
    size_t ob = pbase + (size_t)(c + 1) * SIGC;
    ws_prefix[ob + t] = a0;
    if (t < 2) ws_prefix[ob + 256 + t] = a1;
    p ^= 1;
    __syncthreads();
  }
}

// ---------------- Phase C: local re-scan fused with WMMA GEMM + bias ----------------
__global__ void __launch_bounds__(256) sig_scan_gemm(const float* __restrict__ inp,
                                                     const float* __restrict__ W,
                                                     const float* __restrict__ bias,
                                                     const float* __restrict__ ws_prefix,
                                                     float* __restrict__ out) {
  const int c = blockIdx.x;   // 0..15
  const int b = blockIdx.y;   // 0..63
  extern __shared__ char smem[];
  __bf16* sWt   = (__bf16*)smem;                                   // [128][KPAD] W^T by K
  __bf16* sSig  = (__bf16*)(smem + OUTC * KPAD * 2);               // [16][KPAD] sig tile
  float*  sDx   = (float*)(smem + OUTC * KPAD * 2 + 16 * KPAD * 2);// [127][6]
  float*  sBias = sDx + CHUNK * CCH;                               // [128]
  __shared__ float sA1[2][6];
  __shared__ float sA2[2][36];

  const int t    = threadIdx.x;
  const int lane = t & 31;
  const int wave = t >> 5;
  const int h    = lane >> 4;    // lane half
  const int nloc = lane & 15;
  const int row0 = c * CHUNK;

  // W[n][k] f32 -> LDS bf16 [n][k], K padded with zeros (L2-resident source)
  for (int idx = t; idx < OUTC * KPAD; idx += 256) {
    int n = idx / KPAD, kk = idx % KPAD;
    sWt[idx] = (kk < SIGC) ? f2bf(W[n * SIGC + kk]) : f2bf(0.f);
  }
  for (int idx = t; idx < 16 * KPAD; idx += 256) sSig[idx] = f2bf(0.f); // zero K-pad
  if (t < OUTC) sBias[t] = bias[t];
  for (int idx = t; idx < (CHUNK - 1) * CCH; idx += 256) {
    int l = idx / CCH, ch = idx % CCH;
    int s = row0 + l;                          // dx index, max 2046
    float v;
    if (ch == 0) v = 1.0f / (float)NSTEP;
    else {
      size_t o = ((size_t)b * SEQ + s) * 5 + (ch - 1);
      v = inp[o + 5] - inp[o];
    }
    sDx[idx] = v;
  }

  int l0, i0, j0, k0, l1 = 0, i1 = 0, j1 = 0, k1 = 0;
  decode_elem(t, l0, i0, j0, k0);
  if (t < 2) decode_elem(256 + t, l1, i1, j1, k1);

  const float* pre = &ws_prefix[((size_t)b * NCHUNK + c) * SIGC];
  float a0 = pre[t];
  float a1 = (t < 2) ? pre[256 + t] : 0.f;
  if (t < 6)       sA1[0][t] = a0;
  else if (t < 42) sA2[0][t - 6] = a0;
  __syncthreads();

  int p = 0, step = 0;
  for (int tile = 0; tile < 8; ++tile) {
    // Fill 16 signature rows (bf16) into LDS, advancing the Chen recurrence.
    for (int r = 0; r < 16; ++r) {
      if (tile == 0 && r == 0) {
        sSig[t] = f2bf(a0);                    // row 0 of chunk = carried prefix
        if (t < 2) sSig[256 + t] = f2bf(a1);
      } else {
        const float* dx = &sDx[step * CCH];
        a0 = chen_step(a0, l0, i0, j0, k0, dx, sA1[p], sA2[p]);
        if (t < 2) a1 = chen_step(a1, l1, i1, j1, k1, dx, sA1[p], sA2[p]);
        if (t < 6)       sA1[p ^ 1][t] = a0;
        else if (t < 42) sA2[p ^ 1][t - 6] = a0;
        sSig[r * KPAD + t] = f2bf(a0);
        if (t < 2) sSig[r * KPAD + 256 + t] = f2bf(a1);
        p ^= 1;
        ++step;
      }
      __syncthreads();
    }

    // 16x128 GEMM tile: 8 waves x one 16x16 WMMA tile each, K = 288 (9 steps).
    const int n = wave * 16 + nloc;
    v8f acc;
    {
      float bv = sBias[n];
      #pragma unroll
      for (int r2 = 0; r2 < 8; ++r2) acc[r2] = bv;
    }
    const __bf16* sigRow = &sSig[nloc * KPAD];  // A row m = lane&15
    const __bf16* wRow   = &sWt[n * KPAD];      // B column n, contiguous in K
    #pragma unroll
    for (int kk = 0; kk < KPAD; kk += 32) {
      union { v16bf v; v8bf h8[2]; } A, B;
      A.h8[0] = *(const v8bf*)(sigRow + kk + 8 * h);        // K = kk+8h .. +7
      A.h8[1] = *(const v8bf*)(sigRow + kk + 16 + 8 * h);   // K = kk+16+8h .. +7
      B.h8[0] = *(const v8bf*)(wRow + kk + 8 * h);          // K = kk+8h .. +7
      B.h8[1] = *(const v8bf*)(wRow + kk + 16 + 8 * h);     // K = kk+16+8h .. +7
      acc = __builtin_amdgcn_wmma_f32_16x16x32_bf16(false, A.v, false, B.v,
                                                    (short)0, acc, false, false);
    }
    // D layout: VGPR r -> row r + 8*h, col n
    size_t obase = ((size_t)b * SEQ + row0 + tile * 16);
    #pragma unroll
    for (int r2 = 0; r2 < 8; ++r2) {
      out[(obase + r2 + 8 * h) * OUTC + n] = acc[r2];
    }
    __syncthreads();   // before next tile overwrites sSig
  }
}

extern "C" void kernel_launch(void* const* d_in, const int* in_sizes, int n_in,
                              void* d_out, int out_size, void* d_ws, size_t ws_size,
                              hipStream_t stream) {
  (void)in_sizes; (void)n_in; (void)out_size; (void)ws_size;
  const float* inp  = (const float*)d_in[0];   // [64,2048,5]
  const float* W    = (const float*)d_in[1];   // [128,258]
  const float* bias = (const float*)d_in[2];   // [128]
  float* out = (float*)d_out;                  // [64,2048,128]

  float* ws_total  = (float*)d_ws;                              // 64*15*258 f32
  float* ws_prefix = ws_total + (size_t)NBATCH * NTOT * SIGC;   // 64*16*258 f32

  sig_chunk_totals<<<dim3(NTOT, NBATCH), 256, 0, stream>>>(inp, ws_total);
  sig_prefix<<<dim3(NBATCH), 256, 0, stream>>>(ws_total, ws_prefix);

  size_t shmem = (size_t)OUTC * KPAD * 2   // W bf16
               + (size_t)16 * KPAD * 2     // sig tile bf16
               + (size_t)CHUNK * CCH * 4   // dx
               + (size_t)OUTC * 4;         // bias
  sig_scan_gemm<<<dim3(NCHUNK, NBATCH), 256, shmem, stream>>>(inp, W, bias, ws_prefix, out);
}
